// RNN_Model_36541581754451
// MI455X (gfx1250) — matrix operand order repaired and checked by reading, not compiled, not run on hardware
//
#include <hip/hip_runtime.h>

// ---------------------------------------------------------------- constants
#define S_LEN  1024
#define BATCH  64
#define I_DIM  256
#define H_DIM  512
#define O_DIM  256
#define M_ROWS (S_LEN * BATCH)   // 65536 rows time-major (row = s*64 + b)

typedef __bf16 bf16_t;
typedef __attribute__((ext_vector_type(16))) __bf16   v16bf;
typedef __attribute__((ext_vector_type(2)))  __bf16   v2bf;
typedef __attribute__((ext_vector_type(8)))  float    v8f;
typedef __attribute__((ext_vector_type(8)))  unsigned u32x8;
typedef __attribute__((ext_vector_type(4)))  unsigned u32x4;
typedef __attribute__((ext_vector_type(4)))  int      i32x4;

#if defined(__has_builtin)
#if __has_builtin(__builtin_amdgcn_cvt_pk_bf16_f32)
#define HAVE_CVTPK 1
#endif
#if __has_builtin(__builtin_amdgcn_global_load_async_to_lds_b128)
#define HAVE_ASYNC_LDS 1
#endif
#endif

// two fp32 -> packed bf16 pair (hardware cvt)
__device__ __forceinline__ unsigned pack2bf(float a, float b) {
#ifdef HAVE_CVTPK
  return __builtin_bit_cast(unsigned, __builtin_amdgcn_cvt_pk_bf16_f32(a, b));
#else
  v2bf v = { (bf16_t)a, (bf16_t)b };
  return __builtin_bit_cast(unsigned, v);
#endif
}

__device__ __forceinline__ bf16_t f2bf(float f) { return (bf16_t)f; }

__device__ __forceinline__ void store_bf_pair(bf16_t* p0, bf16_t* p1,
                                              float a, float b) {
  unsigned pk = pack2bf(a, b);
  *p0 = __builtin_bit_cast(bf16_t, (unsigned short)(pk & 0xFFFFu));
  *p1 = __builtin_bit_cast(bf16_t, (unsigned short)(pk >> 16));
}

// ------------------------------------------------------------- WMMA helpers
__device__ __forceinline__ v8f wmma_bf16(v16bf a, v16bf b, v8f c) {
  return __builtin_amdgcn_wmma_f32_16x16x32_bf16(
      false, a, false, b, (short)0, c, false, false);
}

// B tile from pre-swizzled storage: tile (nt,kt) = 32 lanes x 16 bf16,
// lane-contiguous -> one 32B vector load per lane.
__device__ __forceinline__ v16bf load_btile(const bf16_t* __restrict__ Bsw,
                                            int nt, int kt, int ktiles, int lane) {
  const bf16_t* p = Bsw + (((size_t)nt * ktiles + kt) << 9) + (lane << 4);
  return __builtin_bit_cast(v16bf, *(const u32x8*)p);
}

// A tile (16x32 bf16) from row-major bf16 (global or LDS), leading dim lda.
__device__ __forceinline__ v16bf load_atile_bf16(const bf16_t* __restrict__ A,
                                                 size_t lda, int m0, int k0, int lane) {
  int m  = lane & 15;
  int kb = (lane >> 4) << 3;
  const bf16_t* p = A + (size_t)(m0 + m) * lda + k0 + kb;
  u32x4 lo = *(const u32x4*)p;          // K = kb .. kb+7
  u32x4 hi = *(const u32x4*)(p + 16);   // K = kb+16 .. kb+23
  u32x8 c;
  c[0]=lo[0]; c[1]=lo[1]; c[2]=lo[2]; c[3]=lo[3];
  c[4]=hi[0]; c[5]=hi[1]; c[6]=hi[2]; c[7]=hi[3];
  return __builtin_bit_cast(v16bf, c);
}

// ------------------------------------------------- 16B global->LDS copy unit
__device__ __forceinline__ void cp16_to_lds(bf16_t* l, const bf16_t* g) {
#ifdef HAVE_ASYNC_LDS
  __builtin_amdgcn_global_load_async_to_lds_b128(
      (__attribute__((address_space(1))) i32x4*)g,
      (__attribute__((address_space(3))) i32x4*)l, 0, 0);
#else
  *(u32x4*)l = *(const u32x4*)g;
#endif
}
__device__ __forceinline__ void wait_cp() {
#ifdef HAVE_ASYNC_LDS
#if __has_builtin(__builtin_amdgcn_s_wait_asynccnt)
  __builtin_amdgcn_s_wait_asynccnt(0);
#else
  asm volatile("s_wait_asynccnt 0x0" ::: "memory");
#endif
#endif
}

#define LDS_PAD_ROW 520   // 512 + 8 pad -> 4-bank rotate per row

// --------------------------------------------------------- weight swizzler
__global__ void k_swizzle(const float* __restrict__ src, bf16_t* __restrict__ dst,
                          int N, int K) {
  int ktiles = K >> 5;
  int total  = N * K;
  for (int i = blockIdx.x * blockDim.x + threadIdx.x; i < total;
       i += gridDim.x * blockDim.x) {
    int j    = i & 15;
    int lane = (i >> 4) & 31;
    int tile = i >> 9;
    int kt = tile % ktiles;
    int nt = tile / ktiles;
    int n = nt * 16 + (lane & 15);
    int k = kt * 32 + ((lane >> 4) << 3) + ((j >> 3) << 4) + (j & 7);
    dst[i] = f2bf(src[(size_t)n * K + k]);
  }
}

// ------------------------------ x [B,S,I] fp32 -> time-major bf16 [row][I]
__global__ void __launch_bounds__(256) k_cvt_x(const float* __restrict__ x,
                                               unsigned* __restrict__ xbf) {
  int i = blockIdx.x * blockDim.x + threadIdx.x;     // pair index
  if (i >= M_ROWS * I_DIM / 2) return;
  int col2 = i & (I_DIM / 2 - 1);
  int row  = i >> 7;                                 // I_DIM/2 = 128
  int s = row >> 6, b = row & 63;                    // row = s*64 + b
  const float* p = x + ((size_t)b * S_LEN + s) * I_DIM + col2 * 2;
  xbf[i] = pack2bf(p[0], p[1]);
}

// ---------------------------------------------------- unified GEMM template
#define OUT_BF16     0
#define OUT_F32_ROW  1
#define OUT_F32_PERM 2

template <int KDIM, int NCOLS, int NTW, int OMODE>
__global__ void __launch_bounds__(256) k_gemm(const bf16_t* __restrict__ A,
                                              const bf16_t* __restrict__ Bsw,
                                              const float* __restrict__ bias,
                                              void* __restrict__ outv) {
  constexpr int LROW   = KDIM + 8;
  constexpr int CHUNKS = 16 * (KDIM / 8);
  __shared__ bf16_t As[16 * LROW];
  int m0   = blockIdx.x * 16;
  int lane = threadIdx.x & 31;
  int wid  = threadIdx.x >> 5;

  for (int c = threadIdx.x; c < CHUNKS; c += 256) {
    int row = c / (KDIM / 8);
    int off = (c % (KDIM / 8)) * 8;
    cp16_to_lds(&As[row * LROW + off], &A[(size_t)(m0 + row) * KDIM + off]);
  }
  wait_cp();
  __syncthreads();

  v8f acc[NTW] = {};
#pragma unroll
  for (int kt = 0; kt < KDIM / 32; ++kt) {
    v16bf a = load_atile_bf16(As, LROW, 0, kt * 32, lane);
#pragma unroll
    for (int w = 0; w < NTW; ++w)
      acc[w] = wmma_bf16(a, load_btile(Bsw, wid * NTW + w, kt, KDIM / 32, lane),
                         acc[w]);
  }

#pragma unroll
  for (int w = 0; w < NTW; ++w) {
    int n = (wid * NTW + w) * 16 + (lane & 15);
    float bn = bias[n];
    if constexpr (OMODE == OUT_F32_PERM) {
      float* out = (float*)outv;
#pragma unroll
      for (int r = 0; r < 8; ++r) {
        int m = m0 + r + ((lane >> 4) << 3);     // row = s*64 + b
        int s = m >> 6, b = m & 63;
        out[((size_t)b * S_LEN + s) * NCOLS + n] = acc[w][r] + bn;
      }
    } else if constexpr (OMODE == OUT_F32_ROW) {
      float* out = (float*)outv;
#pragma unroll
      for (int r = 0; r < 8; ++r) {
        int m = m0 + r + ((lane >> 4) << 3);
        out[(size_t)m * NCOLS + n] = acc[w][r] + bn;
      }
    } else {
      bf16_t* out = (bf16_t*)outv;
#pragma unroll
      for (int r = 0; r < 8; r += 2) {
        int m = m0 + r + ((lane >> 4) << 3);
        store_bf_pair(&out[(size_t)m * NCOLS + n],
                      &out[(size_t)(m + 1) * NCOLS + n],
                      acc[w][r] + bn, acc[w][r + 1] + bn);
      }
    }
  }
}

// ----------------------------------------- phase 2: persistent recurrence
// 32 blocks x 8 waves; all 8 waves of a block share one row-tile mt.
// Waves 0..127 (blocks 0..15): h_t = tanh(xu_t + h_{t-1}W^T + Wb);
// waves 128..255 (blocks 16..31): o_{t-1} = h_{t-1}V^T + Vb (pipelined).
// Per step: ONE bulk async stage of the 16x512 h strip into LDS (one wait),
// then 16x (ds_load pair -> wmma) at LDS latency. Weights register-resident.
#define RNN_BLOCKS 32
__global__ void __launch_bounds__(256, 1) k_rnn(
    const float* __restrict__ xu,
    const bf16_t* __restrict__ Wsw, const float* __restrict__ Wb,
    const bf16_t* __restrict__ Vsw, const float* __restrict__ Vb,
    bf16_t* __restrict__ hbuf, bf16_t* __restrict__ o,
    unsigned* __restrict__ cnt) {
  __shared__ bf16_t Hs[16 * LDS_PAD_ROW];
  int lane = threadIdx.x & 31;
  int wid  = threadIdx.x >> 5;
  int g    = blockIdx.x * 8 + wid;          // 0..255
  int isV  = g >> 7;
  int idx  = g & 127;
  int mt = idx >> 5, nt = idx & 31;
  int m0 = mt * 16, n0 = nt * 16;
  const bf16_t* Bsw = isV ? Vsw : Wsw;
  float bias = (isV ? Vb : Wb)[n0 + (lane & 15)];

  v16bf breg[16];                           // resident weight strip
#pragma unroll
  for (int kt = 0; kt < 16; ++kt) breg[kt] = load_btile(Bsw, nt, kt, 16, lane);

  const unsigned nb = gridDim.x;
  for (int t = 0; t <= S_LEN; ++t) {
    const bf16_t* hprev = hbuf + (size_t)(t & 1) * BATCH * H_DIM;  // h_{t-1}

    // bulk stage of this block's 16-row strip of h_{t-1} into LDS
    for (int c = threadIdx.x; c < 1024; c += 256) {
      int row = c >> 6;
      int off = (c & 63) << 3;
      cp16_to_lds(&Hs[row * LDS_PAD_ROW + off],
                  &hprev[(size_t)(m0 + row) * H_DIM + off]);
    }
    wait_cp();
    __syncthreads();

    if (!isV && t < S_LEN) {
      v8f acc = {};
#pragma unroll
      for (int kt = 0; kt < 16; ++kt)
        acc = wmma_bf16(load_atile_bf16(Hs, LDS_PAD_ROW, 0, kt * 32, lane),
                        breg[kt], acc);
      bf16_t* hnext = hbuf + (size_t)((t + 1) & 1) * BATCH * H_DIM;
      const float* xurow = xu + (size_t)t * BATCH * H_DIM;
      __builtin_prefetch(xurow + BATCH * H_DIM, 0, 1);
      int n = n0 + (lane & 15);
#pragma unroll
      for (int r = 0; r < 8; r += 2) {
        int m = m0 + r + ((lane >> 4) << 3);
        float v0 = tanhf(acc[r]     + bias + xurow[(size_t)m * H_DIM + n]);
        float v1 = tanhf(acc[r + 1] + bias + xurow[(size_t)(m + 1) * H_DIM + n]);
        store_bf_pair(&hnext[(size_t)m * H_DIM + n],
                      &hnext[(size_t)(m + 1) * H_DIM + n], v0, v1);
      }
    } else if (isV && t > 0) {
      v8f acc = {};
#pragma unroll
      for (int kt = 0; kt < 16; ++kt)
        acc = wmma_bf16(load_atile_bf16(Hs, LDS_PAD_ROW, 0, kt * 32, lane),
                        breg[kt], acc);
      bf16_t* orow = o + (size_t)(t - 1) * BATCH * H_DIM;
      int n = n0 + (lane & 15);
#pragma unroll
      for (int r = 0; r < 8; r += 2) {
        int m = m0 + r + ((lane >> 4) << 3);
        store_bf_pair(&orow[(size_t)m * H_DIM + n],
                      &orow[(size_t)(m + 1) * H_DIM + n],
                      acc[r] + bias, acc[r + 1] + bias);
      }
    }

    // device-wide barrier: monotone counter, target = (t+1)*nblocks
    __syncthreads();
    if (threadIdx.x == 0) {
      __threadfence();
      atomicAdd(cnt, 1u);
      unsigned target = (unsigned)(t + 1) * nb;
      while (atomicAdd(cnt, 0u) < target) __builtin_amdgcn_s_sleep(1);
    }
    __syncthreads();
    __threadfence();
  }
}

// ----------------------------------------------------------------- launch
extern "C" void kernel_launch(void* const* d_in, const int* in_sizes, int n_in,
                              void* d_out, int out_size, void* d_ws, size_t ws_size,
                              hipStream_t stream) {
  const float* x     = (const float*)d_in[0];
  const float* U_w   = (const float*)d_in[1];
  const float* U_b   = (const float*)d_in[2];
  const float* W_w   = (const float*)d_in[3];
  const float* W_b   = (const float*)d_in[4];
  const float* V_w   = (const float*)d_in[5];
  const float* V_b   = (const float*)d_in[6];
  const float* fc1_w = (const float*)d_in[7];
  const float* fc1_b = (const float*)d_in[8];
  const float* fc2_w = (const float*)d_in[9];
  const float* fc2_b = (const float*)d_in[10];
  const float* fc3_w = (const float*)d_in[11];
  const float* fc3_b = (const float*)d_in[12];

  char* ws = (char*)d_ws;
  float*  xu = (float*)(ws);                           // 128MB, dead after k_rnn
  bf16_t* y1 = (bf16_t*)(ws);                          // reuse (64MB)
  bf16_t* y2 = (bf16_t*)(ws + ((size_t)64 << 20));     // reuse (64MB)
  bf16_t* o  = (bf16_t*)(ws + ((size_t)128 << 20));    // 64MB
  char* wb = ws + ((size_t)192 << 20);
  bf16_t* Usw  = (bf16_t*)(wb);
  bf16_t* Wsw  = (bf16_t*)(wb + 262144);
  bf16_t* Vsw  = (bf16_t*)(wb + 786432);
  bf16_t* f1sw = (bf16_t*)(wb + 1310720);
  bf16_t* f2sw = (bf16_t*)(wb + 1835008);
  bf16_t* f3sw = (bf16_t*)(wb + 2359296);
  bf16_t* hbuf = (bf16_t*)(wb + 2621440);
  unsigned* cnt = (unsigned*)(wb + 2621440 + 131072);
  bf16_t* xbf  = (bf16_t*)(ws + ((size_t)196 << 20));  // 32MB, dead after proj

  (void)hipMemsetAsync(cnt, 0, 256, stream);
  (void)hipMemsetAsync(hbuf, 0, (size_t)BATCH * H_DIM * sizeof(bf16_t), stream);

  k_swizzle<<<512, 256, 0, stream>>>(U_w,   Usw,  H_DIM, I_DIM);
  k_swizzle<<<512, 256, 0, stream>>>(W_w,   Wsw,  H_DIM, H_DIM);
  k_swizzle<<<512, 256, 0, stream>>>(V_w,   Vsw,  H_DIM, H_DIM);
  k_swizzle<<<512, 256, 0, stream>>>(fc1_w, f1sw, H_DIM, H_DIM);
  k_swizzle<<<512, 256, 0, stream>>>(fc2_w, f2sw, H_DIM, H_DIM);
  k_swizzle<<<512, 256, 0, stream>>>(fc3_w, f3sw, O_DIM, H_DIM);

  // convert + transpose x once; input projection becomes a pure bf16 GEMM
  k_cvt_x<<<M_ROWS * I_DIM / 2 / 256, 256, 0, stream>>>(x, (unsigned*)xbf);
  k_gemm<I_DIM, H_DIM, 4, OUT_F32_ROW><<<4096, 256, 0, stream>>>(xbf, Usw, U_b, xu);

  k_rnn<<<RNN_BLOCKS, 256, 0, stream>>>(xu, Wsw, W_b, Vsw, V_b, hbuf, o, cnt);

  k_gemm<H_DIM, H_DIM, 4, OUT_BF16    ><<<4096, 256, 0, stream>>>(o,  f1sw, fc1_b, y1);
  k_gemm<H_DIM, H_DIM, 4, OUT_BF16    ><<<4096, 256, 0, stream>>>(y1, f2sw, fc2_b, y2);
  k_gemm<H_DIM, O_DIM, 2, OUT_F32_PERM><<<4096, 256, 0, stream>>>(y2, f3sw, fc3_b, d_out);
}
